// ada_e_conv_layer_50706383897209
// MI455X (gfx1250) — compile-verified, matrix operand order checked by reading, not run on hardware
//
#include <hip/hip_runtime.h>

#define NNODES 100000
#define NEDGES 1600000
#define FEATH  64
#define FEAT   128

typedef __attribute__((ext_vector_type(16))) __bf16 v16bf;
typedef __attribute__((ext_vector_type(8)))  float  v8f;

__device__ __forceinline__ unsigned short f2bf(float f) {
    unsigned u = __float_as_uint(f);
    u += 0x7FFFu + ((u >> 16) & 1u);   // round-to-nearest-even
    return (unsigned short)(u >> 16);
}

// ---------------- CSR build ----------------

__global__ void zero_u32(unsigned int* __restrict__ p, int n) {
    int i = blockIdx.x * blockDim.x + threadIdx.x;
    if (i < n) p[i] = 0u;
}

__global__ void hist_kernel(const int* __restrict__ rows, unsigned int* __restrict__ cnt, int e) {
    int i = blockIdx.x * blockDim.x + threadIdx.x;
    if (i < e) atomicAdd(&cnt[rows[i]], 1u);
}

// single-workgroup chunked exclusive scan over N=100000 (98 chunks of 1024)
__global__ void scan_kernel(const unsigned int* __restrict__ cnt, unsigned int* __restrict__ off, int n) {
    __shared__ unsigned int buf[1024];
    __shared__ unsigned int carry;
    int tid = threadIdx.x;
    if (tid == 0) carry = 0u;
    __syncthreads();
    for (int base = 0; base < n; base += 1024) {
        int i = base + tid;
        unsigned int v = (i < n) ? cnt[i] : 0u;
        buf[tid] = v;
        __syncthreads();
        for (int s = 1; s < 1024; s <<= 1) {
            unsigned int t = (tid >= s) ? buf[tid - s] : 0u;
            __syncthreads();
            buf[tid] += t;
            __syncthreads();
        }
        unsigned int incl  = buf[tid];
        unsigned int total = buf[1023];
        unsigned int excl  = incl - v + carry;
        if (i < n) off[i] = excl;
        __syncthreads();
        if (tid == 0) carry += total;
        __syncthreads();
    }
    if (tid == 0) off[n] = carry;
}

__global__ void scatter_kernel(const int* __restrict__ rows, const int* __restrict__ cols,
                               const float* __restrict__ vals,
                               const unsigned int* __restrict__ off, unsigned int* __restrict__ cur,
                               unsigned int* __restrict__ ecol, float* __restrict__ eval, int e) {
    int i = blockIdx.x * blockDim.x + threadIdx.x;
    if (i < e) {
        int r = rows[i];
        unsigned int p = off[r] + atomicAdd(&cur[r], 1u);
        ecol[p] = (unsigned int)cols[i];
        eval[p] = vals[i];
    }
}

// ---------------- gather SpMM: one wave per node, no output atomics ----------------

__global__ void gather_kernel(const float* __restrict__ x, const unsigned int* __restrict__ off,
                              const unsigned int* __restrict__ ecol, const float* __restrict__ eval,
                              float* __restrict__ out, int half) {
    int wave = (blockIdx.x * blockDim.x + threadIdx.x) >> 5;
    int lane = threadIdx.x & 31;
    if (wave >= NNODES) return;
    unsigned int beg = off[wave], end = off[wave + 1];
    int fb = half * FEATH + lane * 2;
    float ax = 0.f, ay = 0.f;
    for (unsigned int j = beg; j < end; ++j) {
        unsigned int c = ecol[j];   // broadcast across the wave via L0
        float v = eval[j];
        const float2 xv = *reinterpret_cast<const float2*>(x + (size_t)c * FEAT + fb);
        ax = fmaf(v, xv.x, ax);
        ay = fmaf(v, xv.y, ay);
    }
    float2 r; r.x = ax; r.y = ay;
    *reinterpret_cast<float2*>(out + (size_t)wave * FEAT + fb) = r;  // every element written
}

// ---------------- W -> bf16 ----------------

__global__ void cvt_w(const float* __restrict__ W, unsigned short* __restrict__ Wb, int n) {
    int i = blockIdx.x * blockDim.x + threadIdx.x;
    if (i < n) Wb[i] = f2bf(W[i]);
}

// ---------------- in-place WMMA GEMM: out = hidden @ W^T + b ----------------

union U16 {
    v16bf v;
    unsigned short h[16];
    unsigned int u[8];
};

__global__ void gemm_wmma(float* __restrict__ hid_out,              // hidden in, result out (in place)
                          const unsigned short* __restrict__ Wb,    // W bf16, row-major (128x128)
                          const float* __restrict__ bias) {
    int wave = (blockIdx.x * blockDim.x + threadIdx.x) >> 5;
    int lane = threadIdx.x & 31;
    const int ntiles = NNODES / 16;   // 6250
    if (wave >= ntiles) return;
    int m  = lane & 15;
    int hi = lane >> 4;
    size_t rowbase = (size_t)wave * 16;

    // Load this wave's full A block (16 rows x 128 K) and convert to bf16.
    // ISA 16-bit A layout: lanes 0-15 hold K {ks*32+0..7, ks*32+16..23},
    // lanes 16-31 hold K {ks*32+8..15, ks*32+24..31}.
    U16 a[4];
    const float* arow = hid_out + (rowbase + m) * FEAT;
    #pragma unroll
    for (int ks = 0; ks < 4; ++ks) {
        int kb0 = ks * 32 + hi * 8;
        int kb1 = ks * 32 + 16 + hi * 8;
        float4 f0 = *reinterpret_cast<const float4*>(arow + kb0);
        float4 f1 = *reinterpret_cast<const float4*>(arow + kb0 + 4);
        float4 f2 = *reinterpret_cast<const float4*>(arow + kb1);
        float4 f3 = *reinterpret_cast<const float4*>(arow + kb1 + 4);
        a[ks].h[0]  = f2bf(f0.x); a[ks].h[1]  = f2bf(f0.y);
        a[ks].h[2]  = f2bf(f0.z); a[ks].h[3]  = f2bf(f0.w);
        a[ks].h[4]  = f2bf(f1.x); a[ks].h[5]  = f2bf(f1.y);
        a[ks].h[6]  = f2bf(f1.z); a[ks].h[7]  = f2bf(f1.w);
        a[ks].h[8]  = f2bf(f2.x); a[ks].h[9]  = f2bf(f2.y);
        a[ks].h[10] = f2bf(f2.z); a[ks].h[11] = f2bf(f2.w);
        a[ks].h[12] = f2bf(f3.x); a[ks].h[13] = f2bf(f3.y);
        a[ks].h[14] = f2bf(f3.z); a[ks].h[15] = f2bf(f3.w);
    }

    #pragma unroll
    for (int ct = 0; ct < 8; ++ct) {
        int col0 = ct * 16;
        float bb = bias[col0 + m];     // column-n bias broadcast into C
        v8f c;
        #pragma unroll
        for (int i = 0; i < 8; ++i) c[i] = bb;

        const unsigned short* wrow = Wb + (size_t)(col0 + m) * FEAT;  // B[k, n=col0+m] = W[col0+m, k]
        #pragma unroll
        for (int ks = 0; ks < 4; ++ks) {
            int kb0 = ks * 32 + hi * 8;
            int kb1 = ks * 32 + 16 + hi * 8;
            uint4 q0 = *reinterpret_cast<const uint4*>(wrow + kb0);   // 8 bf16 = 16B
            uint4 q1 = *reinterpret_cast<const uint4*>(wrow + kb1);
            U16 bm;
            bm.u[0] = q0.x; bm.u[1] = q0.y; bm.u[2] = q0.z; bm.u[3] = q0.w;
            bm.u[4] = q1.x; bm.u[5] = q1.y; bm.u[6] = q1.z; bm.u[7] = q1.w;
            c = __builtin_amdgcn_wmma_f32_16x16x32_bf16(
                    false, a[ks].v, false, bm.v, (short)0, c, false, false);
        }

        // C/D layout: VGPR r -> M = r + 8*hi, N = m
        float* orow = hid_out + (rowbase + hi * 8) * FEAT + col0 + m;
        #pragma unroll
        for (int r = 0; r < 8; ++r)
            orow[(size_t)r * FEAT] = c[r];
    }
}

// ---------------- host ----------------

extern "C" void kernel_launch(void* const* d_in, const int* in_sizes, int n_in,
                              void* d_out, int out_size, void* d_ws, size_t ws_size,
                              hipStream_t stream) {
    const float* x  = (const float*)d_in[0];
    const int*  ei1 = (const int*)d_in[1];
    const float* v1 = (const float*)d_in[2];
    const int*  ei2 = (const int*)d_in[3];
    const float* v2 = (const float*)d_in[4];
    const float* W  = (const float*)d_in[5];
    const float* b  = (const float*)d_in[6];
    float* out = (float*)d_out;

    char* ws = (char*)d_ws;
    auto alloc = [&](size_t bytes) -> char* {
        char* p = ws;
        ws += (bytes + 255) & ~(size_t)255;
        return p;
    };
    unsigned int* cnt   = (unsigned int*)alloc((size_t)NNODES * 4);
    unsigned int* cur   = (unsigned int*)alloc((size_t)NNODES * 4);
    unsigned int* off   = (unsigned int*)alloc((size_t)(NNODES + 1) * 4);
    unsigned int* ecol  = (unsigned int*)alloc((size_t)NEDGES * 4);
    float*        eval  = (float*)alloc((size_t)NEDGES * 4);
    unsigned short* Wb  = (unsigned short*)alloc((size_t)FEAT * FEAT * 2);

    cvt_w<<<(FEAT * FEAT + 255) / 256, 256, 0, stream>>>(W, Wb, FEAT * FEAT);

    const int*   eis[2] = { ei1, ei2 };
    const float* vls[2] = { v1, v2 };
    for (int s = 0; s < 2; ++s) {
        const int* rows = eis[s];
        const int* cols = eis[s] + NEDGES;
        zero_u32<<<(NNODES + 255) / 256, 256, 0, stream>>>(cnt, NNODES);
        zero_u32<<<(NNODES + 255) / 256, 256, 0, stream>>>(cur, NNODES);
        hist_kernel<<<(NEDGES + 255) / 256, 256, 0, stream>>>(rows, cnt, NEDGES);
        scan_kernel<<<1, 1024, 0, stream>>>(cnt, off, NNODES);
        scatter_kernel<<<(NEDGES + 255) / 256, 256, 0, stream>>>(rows, cols, vls[s], off, cur,
                                                                 ecol, eval, NEDGES);
        gather_kernel<<<((size_t)NNODES * 32 + 255) / 256, 256, 0, stream>>>(x, off, ecol, eval,
                                                                             out, s);
    }

    int gemm_threads = (NNODES / 16) * 32;  // one wave per 16-row tile
    gemm_wmma<<<(gemm_threads + 255) / 256, 256, 0, stream>>>(out, Wb, b);
}
